// Mamba2Model_26525718020396
// MI455X (gfx1250) — compile-verified, matrix operand order checked by reading, not run on hardware
//
#include <hip/hip_runtime.h>
#include <hip/hip_bf16.h>
#include <cstdint>

// ---- model constants ----
static constexpr int kHID    = 1024;
static constexpr int kINTER  = 2048;
static constexpr int kHEADS  = 32;
static constexpr int kHDIM   = 64;
static constexpr int kNSTATE = 128;
static constexpr int kKCONV  = 4;
static constexpr int kCHUNK  = 128;
static constexpr int kNLAYERS= 4;
static constexpr int kCONVD  = kINTER + 2*kNSTATE;            // 2304
static constexpr int kPROJD  = kINTER + kCONVD + kHEADS;      // 4384
static constexpr int kNB     = 2;
static constexpr int kT      = 2048;
static constexpr int kM      = kNB * kT;                      // 4096
static constexpr int kNC     = kT / kCHUNK;                   // 16
static constexpr float kEPS  = 1e-5f;
static constexpr int kLDT    = 136;                            // LDS row stride (halves), 16B aligned

typedef __attribute__((ext_vector_type(16))) __bf16 v16bf;
typedef __attribute__((ext_vector_type(8)))  float  v8f;

union FragBF { uint4 q[2]; v16bf v; };

__device__ __forceinline__ unsigned short f2bf(float f) {
    union { float f; unsigned u; } v; v.f = f;
    unsigned u = v.u;
    u += 0x7fffu + ((u >> 16) & 1u);           // round-to-nearest-even
    return (unsigned short)(u >> 16);
}
__device__ __forceinline__ float bf2f(unsigned short s) {
    union { unsigned u; float f; } v; v.u = ((unsigned)s) << 16; return v.f;
}

// ---------------------------------------------------------------------------
// Generic (optionally batched) bf16 GEMM:  C[m,n] = sum_k A[m,k] * B[n,k]
// One wave per 16x16 output tile, K in steps of 32 via v_wmma_f32_16x16x32_bf16
// ---------------------------------------------------------------------------
__global__ __launch_bounds__(256)
void k_gemm_bf16(const unsigned short* __restrict__ A,
                 const unsigned short* __restrict__ B,
                 float* __restrict__ C,
                 int M, int N, int K, int batches,
                 long long sA, long long sB, long long sC)
{
    int gw   = (int)((blockIdx.x * blockDim.x + threadIdx.x) >> 5);
    int lane = threadIdx.x & 31;
    int ntn = N >> 4, ntm = M >> 4;
    long long tilesPB = (long long)ntm * ntn;
    long long total   = tilesPB * batches;
    if (gw >= total) return;
    int bi = (int)(gw / tilesPB);
    int t  = (int)(gw - (long long)bi * tilesPB);
    int tm = t / ntn, tn = t - tm * ntn;
    int r  = lane & 15;
    int kb = (lane >> 4) << 3;

    const unsigned short* Arow = A + (long long)bi * sA + (long long)(tm*16 + r) * K;
    const unsigned short* Brow = B + (long long)bi * sB + (long long)(tn*16 + r) * K;

    v8f acc = {};
    for (int kk = 0; kk < K; kk += 32) {
        FragBF fa, fb;
        fa.q[0] = *reinterpret_cast<const uint4*>(Arow + kk + kb);
        fa.q[1] = *reinterpret_cast<const uint4*>(Arow + kk + kb + 16);
        fb.q[0] = *reinterpret_cast<const uint4*>(Brow + kk + kb);
        fb.q[1] = *reinterpret_cast<const uint4*>(Brow + kk + kb + 16);
        __builtin_prefetch(Arow + kk + 64, 0, 1);   // global_prefetch_b8
        __builtin_prefetch(Brow + kk + 64, 0, 1);
        acc = __builtin_amdgcn_wmma_f32_16x16x32_bf16(false, fa.v, false, fb.v,
                                                      (short)0, acc, false, false);
    }
    int mo = (lane >> 4) << 3;
#pragma unroll
    for (int j = 0; j < 8; ++j)
        C[(long long)bi * sC + (long long)(tm*16 + j + mo) * N + tn*16 + r] = acc[j];
}

// ---------------------------------------------------------------------------
// Embedding gather
// ---------------------------------------------------------------------------
__global__ void k_embed(const int* __restrict__ ids, const float* __restrict__ emb,
                        float* __restrict__ hs)
{
    int m = blockIdx.x;
    int id = ids[m];
    for (int c = threadIdx.x; c < kHID; c += blockDim.x)
        hs[(long long)m * kHID + c] = emb[(long long)id * kHID + c];
}

// ---------------------------------------------------------------------------
// residual (+=) and rmsnorm -> bf16
// ---------------------------------------------------------------------------
__global__ __launch_bounds__(256)
void k_addnorm(const float* __restrict__ hs, float* __restrict__ res,
               const float* __restrict__ w, unsigned short* __restrict__ outb,
               int first)
{
    int m = blockIdx.x, tid = threadIdx.x;
    __shared__ float red[256];
    float v[4]; float acc = 0.f;
#pragma unroll
    for (int i = 0; i < 4; ++i) {
        int c = tid + i*256;
        float h = hs[(long long)m*kHID + c];
        float nr = first ? h : h + res[(long long)m*kHID + c];
        res[(long long)m*kHID + c] = nr;
        v[i] = nr; acc += nr*nr;
    }
    red[tid] = acc; __syncthreads();
    for (int s = 128; s > 0; s >>= 1) { if (tid < s) red[tid] += red[tid+s]; __syncthreads(); }
    float rs = rsqrtf(red[0] / (float)kHID + kEPS);
#pragma unroll
    for (int i = 0; i < 4; ++i) {
        int c = tid + i*256;
        outb[(long long)m*kHID + c] = f2bf(v[i]*rs*w[c]);
    }
}

// ---------------------------------------------------------------------------
// fp32 -> bf16 weight convert
// ---------------------------------------------------------------------------
__global__ void k_convert(const float* __restrict__ in, unsigned short* __restrict__ out,
                          long long n)
{
    long long i = (long long)blockIdx.x * blockDim.x + threadIdx.x;
    if (i < n) out[i] = f2bf(in[i]);
}

// ---------------------------------------------------------------------------
// causal conv1d (K=4) + SiLU + split into x / B / C (bf16)
// ---------------------------------------------------------------------------
__global__ void k_conv(const float* __restrict__ proj,
                       const float* __restrict__ cw, const float* __restrict__ cb,
                       unsigned short* __restrict__ xb,
                       unsigned short* __restrict__ Bb,
                       unsigned short* __restrict__ Cb)
{
    long long idx = (long long)blockIdx.x * blockDim.x + threadIdx.x;
    if (idx >= (long long)kM * kCONVD) return;
    int ch = (int)(idx % kCONVD);
    int m  = (int)(idx / kCONVD);
    int b = m / kT, t = m - b*kT;
    float s = cb[ch];
#pragma unroll
    for (int k = 0; k < kKCONV; ++k) {
        int tt = t - (kKCONV-1) + k;
        if (tt >= 0)
            s += proj[(long long)(b*kT + tt)*kPROJD + kINTER + ch] * cw[ch*kKCONV + k];
    }
    s = s / (1.f + __expf(-s));        // SiLU
    if (ch < kINTER)                   xb[(long long)m*kINTER + ch] = f2bf(s);
    else if (ch < kINTER + kNSTATE)    Bb[(long long)m*kNSTATE + (ch - kINTER)] = f2bf(s);
    else                               Cb[(long long)m*kNSTATE + (ch - kINTER - kNSTATE)] = f2bf(s);
}

// dt = softplus(dt_raw + dt_bias)
__global__ void k_dt(const float* __restrict__ proj, const float* __restrict__ dtb,
                     float* __restrict__ dt)
{
    int idx = blockIdx.x * blockDim.x + threadIdx.x;
    if (idx >= kM * kHEADS) return;
    int h = idx % kHEADS, m = idx / kHEADS;
    float v = proj[(long long)m*kPROJD + kINTER + kCONVD + h] + dtb[h];
    dt[idx] = (v > 20.f) ? v : log1pf(__expf(v));
}

// ---------------------------------------------------------------------------
// SSD intra-chunk: per (b,chunk,head) — cumsum scan, build masked-decay M in
// LDS, y_intra = M @ x  (WMMA), then chunk state S = (x*w)^T @ B  (WMMA)
// ---------------------------------------------------------------------------
__global__ __launch_bounds__(128)
void k_ssd_intra(const float* __restrict__ CB,
                 const unsigned short* __restrict__ xb,
                 const unsigned short* __restrict__ Bb,
                 const float* __restrict__ dt,
                 const float* __restrict__ A_log,
                 float* __restrict__ acumG, float* __restrict__ decG,
                 float* __restrict__ S, float* __restrict__ y)
{
    int blk = blockIdx.x;
    int h   = blk % kHEADS;
    int bc  = blk / kHEADS;                 // b*kNC + c
    int tid = threadIdx.x;
    int w = tid >> 5, lane = tid & 31;
    int r = lane & 15;
    int kb = (lane >> 4) << 3, mo = (lane >> 4) << 3;
    int gRow0 = bc * kCHUNK;                // == b*kT + c*kCHUNK

    __shared__ __align__(16) unsigned short Mld[kCHUNK * kLDT]; // 34816 B
    __shared__ __align__(16) unsigned short xT [kHDIM * kLDT];  // 17408 B
    __shared__ float s_sc[kCHUNK];
    __shared__ float s_dt[kCHUNK];

    // ---- inclusive scan of a = dt * A over the chunk ----
    float Ah  = -__expf(A_log[h]);
    float dtv = dt[(long long)(gRow0 + tid)*kHEADS + h];
    s_dt[tid] = dtv;
    s_sc[tid] = dtv * Ah;
    __syncthreads();
    for (int off = 1; off < kCHUNK; off <<= 1) {
        float val = s_sc[tid];
        float add = (tid >= off) ? s_sc[tid - off] : 0.f;
        __syncthreads();
        s_sc[tid] = val + add;
        __syncthreads();
    }
    float al    = s_sc[tid];
    float alast = s_sc[kCHUNK - 1];
    acumG[(long long)blk*kCHUNK + tid] = al;
    if (tid == 0) decG[blk] = __expf(alast);

    // ---- build M[l,s] = CB[l,s]*exp(acum[l]-acum[s])*dt[s]  (lower-tri) ----
    {
        const float* CBrow = CB + (long long)bc*kCHUNK*kCHUNK + (long long)tid*kCHUNK;
        for (int sI = 0; sI < kCHUNK; ++sI) {
            float v = 0.f;
            if (sI <= tid) v = CBrow[sI] * __expf(al - s_sc[sI]) * s_dt[sI];
            Mld[tid*kLDT + sI] = f2bf(v);
        }
        const unsigned short* xr = xb + (long long)(gRow0 + tid)*kINTER + h*kHDIM;
        for (int p = 0; p < kHDIM; ++p) xT[p*kLDT + tid] = xr[p];
    }
    __syncthreads();

    // ---- GEMM1: y_intra(128x64) = M(128x128) @ x(128x64) ----
#pragma unroll 1
    for (int i = 0; i < 8; ++i) {
        int idx = w*8 + i;
        int tm = idx >> 2, tn = idx & 3;
        v8f acc = {};
#pragma unroll
        for (int kk = 0; kk < kCHUNK; kk += 32) {
            FragBF fa, fb;
            const unsigned short* ap = &Mld[(tm*16 + r)*kLDT + kk + kb];
            fa.q[0] = *reinterpret_cast<const uint4*>(ap);
            fa.q[1] = *reinterpret_cast<const uint4*>(ap + 16);
            const unsigned short* bp = &xT[(tn*16 + r)*kLDT + kk + kb];
            fb.q[0] = *reinterpret_cast<const uint4*>(bp);
            fb.q[1] = *reinterpret_cast<const uint4*>(bp + 16);
            acc = __builtin_amdgcn_wmma_f32_16x16x32_bf16(false, fa.v, false, fb.v,
                                                          (short)0, acc, false, false);
        }
#pragma unroll
        for (int j = 0; j < 8; ++j) {
            int l = tm*16 + j + mo;
            y[(long long)(gRow0 + l)*kINTER + h*kHDIM + tn*16 + r] = acc[j];
        }
    }
    __syncthreads();

    // ---- reload LDS: Mld <- B^T (n x l); xT <- xT * w[l] in place ----
    {
        const unsigned short* Br = Bb + (long long)(gRow0 + tid)*kNSTATE;
        for (int n = 0; n < kNSTATE; ++n) Mld[n*kLDT + tid] = Br[n];
        float wl = __expf(alast - al) * s_dt[tid];
        for (int p = 0; p < kHDIM; ++p) {
            unsigned short uv = xT[p*kLDT + tid];
            xT[p*kLDT + tid] = f2bf(bf2f(uv) * wl);
        }
    }
    __syncthreads();

    // ---- GEMM2: S(64x128) = (x*w)^T (64x128_l) @ B (128_l x 128_n) ----
    float* Sb = S + (long long)blk * (kHDIM * kNSTATE);
#pragma unroll 1
    for (int i = 0; i < 8; ++i) {
        int idx = w*8 + i;
        int tm = idx & 3, tn = idx >> 2;   // tm over p(4), tn over n(8)
        v8f acc = {};
#pragma unroll
        for (int kk = 0; kk < kCHUNK; kk += 32) {
            FragBF fa, fb;
            const unsigned short* ap = &xT[(tm*16 + r)*kLDT + kk + kb];
            fa.q[0] = *reinterpret_cast<const uint4*>(ap);
            fa.q[1] = *reinterpret_cast<const uint4*>(ap + 16);
            const unsigned short* bp = &Mld[(tn*16 + r)*kLDT + kk + kb];
            fb.q[0] = *reinterpret_cast<const uint4*>(bp);
            fb.q[1] = *reinterpret_cast<const uint4*>(bp + 16);
            acc = __builtin_amdgcn_wmma_f32_16x16x32_bf16(false, fa.v, false, fb.v,
                                                          (short)0, acc, false, false);
        }
#pragma unroll
        for (int j = 0; j < 8; ++j)
            Sb[(long long)(tm*16 + j + mo)*kNSTATE + tn*16 + r] = acc[j];
    }
}

// ---------------------------------------------------------------------------
// sequential inter-chunk state scan (16 chunks) per (b,h)
// ---------------------------------------------------------------------------
__global__ __launch_bounds__(256)
void k_scan(const float* __restrict__ S, const float* __restrict__ dec,
            unsigned short* __restrict__ Sprev)
{
    int h = blockIdx.x, b = blockIdx.y, tid = threadIdx.x;
    const int PN = kHDIM * kNSTATE;        // 8192
    float run[PN/256];
#pragma unroll
    for (int i = 0; i < PN/256; ++i) run[i] = 0.f;
    for (int c = 0; c < kNC; ++c) {
        int blk = (b*kNC + c)*kHEADS + h;
        long long base = (long long)blk * PN;
        float d = dec[blk];
#pragma unroll
        for (int i = 0; i < PN/256; ++i) {
            int e = tid + i*256;
            Sprev[base + e] = f2bf(run[i]);
            run[i] = d*run[i] + S[base + e];
        }
    }
}

// ---------------------------------------------------------------------------
// SSD inter-chunk: y += exp(acum[l]) * (C @ Sprev^T)  + D * x   (WMMA)
// ---------------------------------------------------------------------------
__global__ __launch_bounds__(128)
void k_ssd_inter(const unsigned short* __restrict__ Cb,
                 const unsigned short* __restrict__ Sprev,
                 const float* __restrict__ acumG,
                 const unsigned short* __restrict__ xb,
                 const float* __restrict__ Dp,
                 float* __restrict__ y)
{
    int blk = blockIdx.x;
    int h   = blk % kHEADS;
    int bc  = blk / kHEADS;
    int tid = threadIdx.x;
    int w = tid >> 5, lane = tid & 31;
    int r = lane & 15;
    int kb = (lane >> 4) << 3, mo = (lane >> 4) << 3;
    int gRow0 = bc * kCHUNK;
    float Dh = Dp[h];
    const unsigned short* Sb = Sprev + (long long)blk * (kHDIM * kNSTATE);

#pragma unroll 1
    for (int i = 0; i < 8; ++i) {
        int idx = w*8 + i;
        int tm = idx >> 2, tn = idx & 3;   // tm over l tiles(8), tn over p tiles(4)
        v8f acc = {};
#pragma unroll
        for (int kk = 0; kk < kNSTATE; kk += 32) {
            FragBF fa, fb;
            const unsigned short* ap = Cb + (long long)(gRow0 + tm*16 + r)*kNSTATE + kk + kb;
            fa.q[0] = *reinterpret_cast<const uint4*>(ap);
            fa.q[1] = *reinterpret_cast<const uint4*>(ap + 16);
            const unsigned short* bp = Sb + (long long)(tn*16 + r)*kNSTATE + kk + kb;
            fb.q[0] = *reinterpret_cast<const uint4*>(bp);
            fb.q[1] = *reinterpret_cast<const uint4*>(bp + 16);
            acc = __builtin_amdgcn_wmma_f32_16x16x32_bf16(false, fa.v, false, fb.v,
                                                          (short)0, acc, false, false);
        }
#pragma unroll
        for (int j = 0; j < 8; ++j) {
            int l = tm*16 + j + mo;
            int p = tn*16 + r;
            long long yi = (long long)(gRow0 + l)*kINTER + h*kHDIM + p;
            float e = __expf(acumG[(long long)blk*kCHUNK + l]);
            y[yi] = y[yi] + e*acc[j] + Dh*bf2f(xb[yi]);
        }
    }
}

// ---------------------------------------------------------------------------
// gated rmsnorm: out = rmsnorm(y * silu(z), gw)  -> bf16
// ---------------------------------------------------------------------------
__global__ __launch_bounds__(256)
void k_gatednorm(const float* __restrict__ y, const float* __restrict__ proj,
                 const float* __restrict__ gw, unsigned short* __restrict__ outb)
{
    int m = blockIdx.x, tid = threadIdx.x;
    __shared__ float red[256];
    float v[8]; float acc = 0.f;
#pragma unroll
    for (int i = 0; i < 8; ++i) {
        int c = tid + i*256;
        float z = proj[(long long)m*kPROJD + c];
        float g = y[(long long)m*kINTER + c] * (z / (1.f + __expf(-z)));
        v[i] = g; acc += g*g;
    }
    red[tid] = acc; __syncthreads();
    for (int s = 128; s > 0; s >>= 1) { if (tid < s) red[tid] += red[tid+s]; __syncthreads(); }
    float rs = rsqrtf(red[0] / (float)kINTER + kEPS);
#pragma unroll
    for (int i = 0; i < 8; ++i) {
        int c = tid + i*256;
        outb[(long long)m*kINTER + c] = f2bf(v[i]*rs*gw[c]);
    }
}

// final rmsnorm(hs + residual) -> fp32 output
__global__ __launch_bounds__(256)
void k_final(const float* __restrict__ hs, const float* __restrict__ res,
             const float* __restrict__ w, float* __restrict__ out)
{
    int m = blockIdx.x, tid = threadIdx.x;
    __shared__ float red[256];
    float v[4]; float acc = 0.f;
#pragma unroll
    for (int i = 0; i < 4; ++i) {
        int c = tid + i*256;
        float x = hs[(long long)m*kHID + c] + res[(long long)m*kHID + c];
        v[i] = x; acc += x*x;
    }
    red[tid] = acc; __syncthreads();
    for (int s = 128; s > 0; s >>= 1) { if (tid < s) red[tid] += red[tid+s]; __syncthreads(); }
    float rs = rsqrtf(red[0] / (float)kHID + kEPS);
#pragma unroll
    for (int i = 0; i < 4; ++i) {
        int c = tid + i*256;
        out[(long long)m*kHID + c] = v[i]*rs*w[c];
    }
}

// ---------------------------------------------------------------------------
extern "C" void kernel_launch(void* const* d_in, const int* in_sizes, int n_in,
                              void* d_out, int out_size, void* d_ws, size_t ws_size,
                              hipStream_t stream)
{
    const int*   ids   = (const int*)d_in[0];
    const float* emb   = (const float*)d_in[1];
    const float* ipw   = (const float*)d_in[2];
    const float* cw    = (const float*)d_in[3];
    const float* cb    = (const float*)d_in[4];
    const float* A_log = (const float*)d_in[5];
    const float* Dp    = (const float*)d_in[6];
    const float* dtb   = (const float*)d_in[7];
    const float* nw    = (const float*)d_in[8];
    const float* gnw   = (const float*)d_in[9];
    const float* opw   = (const float*)d_in[10];
    const float* fnw   = (const float*)d_in[11];
    float* out = (float*)d_out;

    char* p = (char*)d_ws;
    auto carve = [&](size_t bytes) -> char* {
        char* r = p; p += (bytes + 255) & ~(size_t)255; return r;
    };
    float* hs   = (float*)carve((size_t)kM*kHID*4);
    float* res  = (float*)carve((size_t)kM*kHID*4);
    unsigned short* xnb = (unsigned short*)carve((size_t)kM*kHID*2);
    float* proj = (float*)carve((size_t)kM*kPROJD*4);
    unsigned short* xb  = (unsigned short*)carve((size_t)kM*kINTER*2);
    unsigned short* Bb  = (unsigned short*)carve((size_t)kM*kNSTATE*2);
    unsigned short* Cbp = (unsigned short*)carve((size_t)kM*kNSTATE*2);
    float* dt   = (float*)carve((size_t)kM*kHEADS*4);
    float* CBf  = (float*)carve((size_t)kNB*kNC*kCHUNK*kCHUNK*4);
    float* acum = (float*)carve((size_t)kNB*kNC*kHEADS*kCHUNK*4);
    float* dec  = (float*)carve((size_t)kNB*kNC*kHEADS*4);
    float* S    = (float*)carve((size_t)kNB*kNC*kHEADS*kHDIM*kNSTATE*4);
    unsigned short* Sprev = (unsigned short*)carve((size_t)kNB*kNC*kHEADS*kHDIM*kNSTATE*2);
    float* y    = (float*)carve((size_t)kM*kINTER*4);
    unsigned short* yb  = (unsigned short*)carve((size_t)kM*kINTER*2);
    unsigned short* wbf = (unsigned short*)carve((size_t)kPROJD*kHID*2);

    k_embed<<<kM, 256, 0, stream>>>(ids, emb, hs);

    const int NBC = kNB * kNC;  // 32 chunk-batches
    for (int L = 0; L < kNLAYERS; ++L) {
        k_addnorm<<<kM, 256, 0, stream>>>(hs, res, nw + (size_t)L*kHID, xnb, L == 0);

        { long long n = (long long)kPROJD*kHID;
          k_convert<<<(unsigned)((n+255)/256), 256, 0, stream>>>(ipw + (size_t)L*n, wbf, n); }
        { long long waves = (long long)(kM/16)*(kPROJD/16);
          k_gemm_bf16<<<(unsigned)((waves+7)/8), 256, 0, stream>>>(
              xnb, wbf, proj, kM, kPROJD, kHID, 1, 0, 0, 0); }

        { long long n = (long long)kM*kCONVD;
          k_conv<<<(unsigned)((n+255)/256), 256, 0, stream>>>(
              proj, cw + (size_t)L*kCONVD*kKCONV, cb + (size_t)L*kCONVD, xb, Bb, Cbp); }
        k_dt<<<(kM*kHEADS+255)/256, 256, 0, stream>>>(proj, dtb + (size_t)L*kHEADS, dt);

        { long long waves = (long long)NBC * (kCHUNK/16) * (kCHUNK/16);
          k_gemm_bf16<<<(unsigned)((waves+7)/8), 256, 0, stream>>>(
              Cbp, Bb, CBf, kCHUNK, kCHUNK, kNSTATE, NBC,
              (long long)kCHUNK*kNSTATE, (long long)kCHUNK*kNSTATE,
              (long long)kCHUNK*kCHUNK); }

        k_ssd_intra<<<NBC*kHEADS, 128, 0, stream>>>(CBf, xb, Bb, dt,
            A_log + (size_t)L*kHEADS, acum, dec, S, y);
        k_scan<<<dim3(kHEADS, kNB), 256, 0, stream>>>(S, dec, Sprev);
        k_ssd_inter<<<NBC*kHEADS, 128, 0, stream>>>(Cbp, Sprev, acum, xb,
            Dp + (size_t)L*kHEADS, y);

        k_gatednorm<<<kM, 256, 0, stream>>>(y, proj, gnw + (size_t)L*kINTER, yb);

        { long long n = (long long)kHID*kINTER;
          k_convert<<<(unsigned)((n+255)/256), 256, 0, stream>>>(opw + (size_t)L*n, wbf, n); }
        { long long waves = (long long)(kM/16)*(kHID/16);
          k_gemm_bf16<<<(unsigned)((waves+7)/8), 256, 0, stream>>>(
              yb, wbf, hs, kM, kHID, kINTER, 1, 0, 0, 0); }
    }
    k_final<<<kM, 256, 0, stream>>>(hs, res, fnw, out);
}
